// AEQMPatchSegmModel_27908697489861
// MI455X (gfx1250) — matrix operand order repaired and checked by reading, not compiled, not run on hardware
//
#include <hip/hip_runtime.h>
#include <hip/hip_fp16.h>

// ---- problem constants ----
#define IMGW    96
#define NPDIM   45
#define NPATCH  2025          // 45*45 patches per image
#define BIMG    16
#define ENC     64
#define NCOMP   128
#define CLS     10

// ---- tiling ----
#define PB      8             // patches per workgroup
#define GPI     254           // ceil(2025/8) groups per image
#define NWAVES  4
#define TPB     128

typedef __attribute__((ext_vector_type(16))) _Float16     v16h;
typedef __attribute__((ext_vector_type(8)))  float        v8f;
typedef __attribute__((ext_vector_type(8)))  unsigned int v8u;

union ABFrag { v8u u; v16h h; };

// ---- workspace byte offsets ----
#define OFF_W1T   0u           // f16 [32][64]
#define OFF_W2T   4096u        // f16 [64][288]
#define OFF_W3T   40960u       // f16 [128][576]
#define OFF_WDT   188416u      // f16 [64][512]
#define OFF_CXT   253952u      // f16 [128][64]
#define OFF_CXN2  270336u      // f32 [128]
#define OFF_YACC  270848u      // f32 [16][128]

// =====================================================================
// Weight prep: transpose to [N][K] f16 (K contiguous), pad conv1 to C=4,
// precompute |c_x|^2, zero y_acc.
// =====================================================================
__global__ void prep_kernel(const float* __restrict__ w1, const float* __restrict__ w2,
                            const float* __restrict__ w3, const float* __restrict__ wd,
                            const float* __restrict__ cx,
                            _Float16* w1T, _Float16* w2T, _Float16* w3T,
                            _Float16* wdT, _Float16* cxT, float* cxn2, float* y_acc)
{
    const int tid = blockIdx.x * blockDim.x + threadIdx.x;
    const int nth = gridDim.x * blockDim.x;

    // conv1: HWIO (3,3,3,32) -> [n][q*4+c], K padded 36->64
    for (int e = tid; e < 32 * 64; e += nth) {
        int n = e >> 6, K = e & 63;
        int q = K >> 2, c = K & 3;
        float v = (q < 9 && c < 3) ? w1[(q * 3 + c) * 32 + n] : 0.0f;
        w1T[n * 64 + K] = (_Float16)v;
    }
    // conv2: (3,3,32,64) -> [n][K], K = q*32+c = flat HWI
    for (int e = tid; e < 64 * 288; e += nth) {
        int n = e / 288, K = e % 288;
        w2T[n * 288 + K] = (_Float16)w2[K * 64 + n];
    }
    // conv3: (3,3,64,128) -> [n][K]
    for (int e = tid; e < 128 * 576; e += nth) {
        int n = e / 576, K = e % 576;
        w3T[n * 576 + K] = (_Float16)w3[K * 128 + n];
    }
    // dense: (512,64) -> [n][512]
    for (int e = tid; e < 64 * 512; e += nth) {
        int n = e >> 9, K = e & 511;
        wdT[n * 512 + K] = (_Float16)wd[K * 64 + n];
    }
    // c_x: (128,64) already [N][K]
    for (int e = tid; e < NCOMP * ENC; e += nth)
        cxT[e] = (_Float16)cx[e];
    // |c_x|^2
    for (int e = tid; e < NCOMP; e += nth) {
        float s = 0.0f;
        for (int j = 0; j < ENC; ++j) { float x = cx[e * ENC + j]; s += x * x; }
        cxn2[e] = s;
    }
    // zero accumulator
    for (int e = tid; e < BIMG * NCOMP; e += nth)
        y_acc[e] = 0.0f;
}

// =====================================================================
// Implicit-im2col conv stage via WMMA f16 (16x16x32), fp32 accumulate.
// k-outer / m-inner: each wave owns a fixed n-tile column; B fragments
// are loaded once per k-step and reused across a chunk of MCH m-tiles
// held in register accumulators.
// actIn  : LDS [PB][W][W][C] f16
// actOut : LDS [PB][OW][OW][Cout] f16  (bias + relu applied)
// wT     : global f16 [Cout][Kpad]
// =====================================================================
template<int C, int W, int KT, int Kpad, int KS, int OW, int Cout, int MT, int NT>
__device__ __forceinline__ void conv_stage(const _Float16* __restrict__ actIn,
                                           _Float16* __restrict__ actOut,
                                           const _Float16* __restrict__ wT,
                                           const float* __restrict__ bias,
                                           int wave, int lane)
{
    constexpr int PPo  = OW * OW;                              // output positions / patch
    constexpr int WPN  = (NWAVES > NT) ? (NWAVES / NT) : 1;    // waves sharing one n-col
    constexpr int NTPW = (NT > NWAVES) ? (NT / NWAVES) : 1;    // n-cols per wave
    constexpr int MTW  = MT / WPN;                             // m-tiles per wave
    constexpr int MCH  = (MTW < 4) ? MTW : 4;                  // m-tile chunk (reg accs)

    const int lm     = lane & 15;
    const int khalf  = (lane & 16) ? 8 : 0;
    const int mhalf  = (lane >> 4) << 3;
    const int ntBase = (NWAVES > NT) ? (wave / WPN) : wave;
    const int mtOff  = (NWAVES > NT) ? (wave % WPN) * MTW : 0;

    for (int g = 0; g < NTPW; ++g) {
        const int nt = ntBase + g * NWAVES;
        const int n  = nt * 16 + lm;
        const _Float16* wrow = wT + n * Kpad + ((lane >> 4) << 3);
        const float bv = bias[n];

        for (int c0 = 0; c0 < MTW; c0 += MCH) {
            v8f acc[MCH];
#pragma unroll
            for (int i = 0; i < MCH; ++i) acc[i] = {};

            for (int ks = 0; ks < KS; ++ks) {
                // ---- B fragment: loaded once, reused across MCH m-tiles ----
                ABFrag bb;
#pragma unroll
                for (int j = 0; j < 8; ++j)
                    bb.u[j] = *(const unsigned*)(wrow + ks * 32 + (j >> 2) * 16 + (j & 3) * 2);

#pragma unroll
                for (int i = 0; i < MCH; ++i) {
                    const int mt    = mtOff + c0 + i;
                    const int m     = mt * 16 + lm;
                    const int patch = m / PPo;
                    const int pos   = m - patch * PPo;
                    const int oy    = pos / OW, ox = pos - (pos / OW) * OW;
                    // ---- A fragment: implicit im2col gather from LDS ----
                    ABFrag a;
#pragma unroll
                    for (int j = 0; j < 8; ++j) {
                        const int K0 = ks * 32 + (j >> 2) * 16 + khalf + (j & 3) * 2;
                        unsigned u = 0u;
                        if (K0 < KT) {
                            const int q  = K0 / C, cc = K0 - q * C;
                            const int ky = q / 3,  kx = q - (q / 3) * 3;
                            const int iy = 2 * oy + ky - 1;
                            const int ix = 2 * ox + kx - 1;
                            if (iy >= 0 && iy < W && ix >= 0 && ix < W)
                                u = *(const unsigned*)(actIn + (((patch * W + iy) * W + ix) * C + cc));
                        }
                        a.u[j] = u;
                    }
                    acc[i] = __builtin_amdgcn_wmma_f32_16x16x32_f16(
                        false, a.h, false, bb.h, (short)0, acc[i], false, false);
                }
            }

            // ---- bias + relu, store f16 to LDS ----
#pragma unroll
            for (int i = 0; i < MCH; ++i) {
                const int mt = mtOff + c0 + i;
#pragma unroll
                for (int r = 0; r < 8; ++r) {
                    const int mm  = mt * 16 + r + mhalf;
                    const int pch = mm / PPo;
                    const int pp  = mm - pch * PPo;
                    const int oy2 = pp / OW, ox2 = pp - (pp / OW) * OW;
                    float v = acc[i][r] + bv;
                    v = v > 0.0f ? v : 0.0f;
                    actOut[((pch * OW + oy2) * OW + ox2) * Cout + n] = (_Float16)v;
                }
            }
        }
    }
}

// =====================================================================
// Fully fused patch pipeline: extract+resize -> conv1/2/3 -> dense ->
// v.c_x GEMM -> RBF epilogue -> atomic accumulate per-image y_w.
// One workgroup = 8 patches, 128 threads (4 waves).
// =====================================================================
__global__ __launch_bounds__(TPB)
void fused_kernel(const float* __restrict__ images,
                  const float* __restrict__ b1, const float* __restrict__ b2,
                  const float* __restrict__ b3, const float* __restrict__ bd,
                  const float* __restrict__ comp_w, const float* __restrict__ sigma_p,
                  const _Float16* __restrict__ w1T, const _Float16* __restrict__ w2T,
                  const _Float16* __restrict__ w3T, const _Float16* __restrict__ wdT,
                  const _Float16* __restrict__ cxT, const float* __restrict__ cxn2,
                  float* __restrict__ y_acc)
{
    const int tid  = threadIdx.x;
    const int lane = tid & 31;
    const int wave = tid >> 5;
    const int b    = blockIdx.y;   // image
    const int gi   = blockIdx.x;   // patch group

    __shared__ __align__(16) _Float16     s_a1[PB * 8 * 8 * 32];   // 32 KB
    __shared__ __align__(16) unsigned char s_u[29760];              // union region
    _Float16* s_in  = (_Float16*)s_u;                               // [PB][16][16][4] 16 KB (dies after conv1)
    _Float16* s_a2  = (_Float16*)s_u;                               // [PB][4][4][64]  16 KB
    _Float16* s_a3  = (_Float16*)(s_u + 16384);                     // [PB][512]        8 KB
    _Float16* s_vh  = (_Float16*)(s_u + 16384 + 8192);              // [PB][64]         1 KB
    float*    s_vcx = (float*)   (s_u + 16384 + 8192 + 1024);       // [PB][128]        4 KB
    float*    s_vn2 = (float*)   (s_u + 16384 + 8192 + 1024 + 4096);// [PB]

    // ---------------- Stage 0: extract 8x8 patch + bilinear resize to 16x16, pad C to 4
    for (int e = tid; e < PB * 256; e += TPB) {
        const int p  = e >> 8, pix = e & 255;
        const int oy = pix >> 4, ox = pix & 15;
        const int gp = gi * PB + p;
        const int pp = gp < NPATCH ? gp : 0;
        const int py = pp / NPDIM, px = pp - (pp / NPDIM) * NPDIM;

        // half-pixel-center bilinear, scale 8->16: src = 0.5*i - 0.25
        const float sy = 0.5f * oy - 0.25f;
        const float sx = 0.5f * ox - 0.25f;
        const int y0 = (int)floorf(sy); const float ty = sy - (float)y0;
        const int x0 = (int)floorf(sx); const float tx = sx - (float)x0;
        const int y0c = y0 < 0 ? 0 : (y0 > 7 ? 7 : y0);
        const int y1c = (y0 + 1) < 0 ? 0 : ((y0 + 1) > 7 ? 7 : y0 + 1);
        const int x0c = x0 < 0 ? 0 : (x0 > 7 ? 7 : x0);
        const int x1c = (x0 + 1) < 0 ? 0 : ((x0 + 1) > 7 ? 7 : x0 + 1);

        const float* base = images + ((size_t)(b * IMGW + py * 2) * IMGW + px * 2) * 3;
#pragma unroll
        for (int c = 0; c < 4; ++c) {
            float v = 0.0f;
            if (c < 3) {
                const float v00 = base[(y0c * IMGW + x0c) * 3 + c];
                const float v01 = base[(y0c * IMGW + x1c) * 3 + c];
                const float v10 = base[(y1c * IMGW + x0c) * 3 + c];
                const float v11 = base[(y1c * IMGW + x1c) * 3 + c];
                v = (1.0f - ty) * ((1.0f - tx) * v00 + tx * v01)
                  +          ty * ((1.0f - tx) * v10 + tx * v11);
            }
            s_in[((p * 16 + oy) * 16 + ox) * 4 + c] = (_Float16)v;
        }
    }
    __syncthreads();

    // ---------------- conv1: 16x16x4 -> 8x8x32, K=36 (pad 64)
    conv_stage<4, 16, 36, 64, 2, 8, 32, /*MT=*/(PB * 64) / 16, /*NT=*/2>(
        s_in, s_a1, w1T, b1, wave, lane);
    __syncthreads();

    // ---------------- conv2: 8x8x32 -> 4x4x64, K=288
    conv_stage<32, 8, 288, 288, 9, 4, 64, /*MT=*/(PB * 16) / 16, /*NT=*/4>(
        s_a1, s_a2, w2T, b2, wave, lane);
    __syncthreads();

    // ---------------- conv3: 4x4x64 -> 2x2x128, K=576
    conv_stage<64, 4, 576, 576, 18, 2, 128, /*MT=*/(PB * 4) / 16, /*NT=*/8>(
        s_a2, s_a3, w3T, b3, wave, lane);
    __syncthreads();

    // ---------------- dense: [PB][512] @ wd(512,64) + bd  -> v [PB][64]
    {
        const int lm    = lane & 15;
        const int khalf = (lane & 16) ? 8 : 0;
        const int patch = lm < PB ? lm : (PB - 1);
        const int nt    = wave;                 // 4 n-tiles over 4 waves
        const int n     = nt * 16 + lm;
        v8f acc = {};
        for (int ks = 0; ks < 16; ++ks) {
            ABFrag a, bb;
#pragma unroll
            for (int j = 0; j < 8; ++j) {
                const int K0 = ks * 32 + (j >> 2) * 16 + khalf + (j & 3) * 2;
                a.u[j] = *(const unsigned*)(s_a3 + patch * 512 + K0);
            }
            const _Float16* bp = wdT + n * 512 + ks * 32 + ((lane >> 4) << 3);
#pragma unroll
            for (int j = 0; j < 8; ++j)
                bb.u[j] = *(const unsigned*)(bp + (j >> 2) * 16 + (j & 3) * 2);
            acc = __builtin_amdgcn_wmma_f32_16x16x32_f16(false, a.h, false, bb.h,
                                                         (short)0, acc, false, false);
        }
        const float bv = bd[n];
#pragma unroll
        for (int r = 0; r < 8; ++r) {
            const int mm = r + ((lane >> 4) << 3);
            if (mm < PB) s_vh[mm * ENC + n] = (_Float16)(acc[r] + bv);
        }
    }
    __syncthreads();

    // ---------------- RBF GEMM: v [PB][64] @ c_x^T (64 x 128) -> s_vcx
    {
        const int lm    = lane & 15;
        const int khalf = (lane & 16) ? 8 : 0;
        const int patch = lm < PB ? lm : (PB - 1);
        for (int t = wave; t < 8; t += NWAVES) {    // 8 n-tiles
            const int n = t * 16 + lm;
            v8f acc = {};
            for (int ks = 0; ks < 2; ++ks) {
                ABFrag a, bb;
#pragma unroll
                for (int j = 0; j < 8; ++j) {
                    const int K0 = ks * 32 + (j >> 2) * 16 + khalf + (j & 3) * 2;
                    a.u[j] = *(const unsigned*)(s_vh + patch * ENC + K0);
                }
                const _Float16* bp = cxT + n * ENC + ks * 32 + ((lane >> 4) << 3);
#pragma unroll
                for (int j = 0; j < 8; ++j)
                    bb.u[j] = *(const unsigned*)(bp + (j >> 2) * 16 + (j & 3) * 2);
                acc = __builtin_amdgcn_wmma_f32_16x16x32_f16(false, a.h, false, bb.h,
                                                             (short)0, acc, false, false);
            }
#pragma unroll
            for (int r = 0; r < 8; ++r) {
                const int mm = r + ((lane >> 4) << 3);
                if (mm < PB) s_vcx[mm * NCOMP + n] = acc[r];
            }
        }
    }
    __syncthreads();

    // ---------------- |v|^2 per patch
    if (tid < PB) {
        float s = 0.0f;
        for (int k = 0; k < ENC; ++k) { float x = (float)s_vh[tid * ENC + k]; s += x * x; }
        s_vn2[tid] = s;
    }
    __syncthreads();

    // ---------------- RBF epilogue + per-image accumulate (128 threads = 128 comps)
    {
        const int k = tid;
        const float sg     = sigma_p[0];
        const float inv2s2 = 1.0f / (2.0f * sg * sg);
        const float cxn    = cxn2[k];
        const float cw     = comp_w[k];
        float acc = 0.0f;
        for (int p = 0; p < PB; ++p) {
            const int gp = gi * PB + p;
            if (gp >= NPATCH) continue;
            float d2 = s_vn2[p] + cxn - 2.0f * s_vcx[p * NCOMP + k];
            d2 = d2 > 0.0f ? d2 : 0.0f;
            const float Kv = __expf(-d2 * inv2s2);
            float ow = cw * Kv * Kv;
            ow = ow > 1e-10f ? ow : 1e-10f;
            acc += ow * (1.0f / (float)NPATCH);
        }
        atomicAdd(&y_acc[b * NCOMP + k], acc);
    }
}

// =====================================================================
// Final: normalize y_w, project through normalized c_y^2 -> probs (16x10)
// =====================================================================
__global__ __launch_bounds__(192)
void finish_kernel(const float* __restrict__ cy, const float* __restrict__ y_acc,
                   float* __restrict__ out)
{
    __shared__ float cyn[NCOMP];
    const int t = threadIdx.x;
    if (t < NCOMP) {
        float s = 0.0f;
        for (int i = 0; i < CLS; ++i) { float q = cy[t * CLS + i]; s += q * q; }
        cyn[t] = s;
    }
    __syncthreads();
    if (t < BIMG * CLS) {
        const int bimg = t / CLS, i = t - (t / CLS) * CLS;
        float S = 0.0f, s = 0.0f;
        for (int k = 0; k < NCOMP; ++k) {
            const float w = y_acc[bimg * NCOMP + k];
            S += w;
            const float q = cy[k * CLS + i];
            s += w * (q * q) / cyn[k];
        }
        out[bimg * CLS + i] = s / S;
    }
}

// =====================================================================
extern "C" void kernel_launch(void* const* d_in, const int* in_sizes, int n_in,
                              void* d_out, int out_size, void* d_ws, size_t ws_size,
                              hipStream_t stream)
{
    (void)in_sizes; (void)n_in; (void)out_size; (void)ws_size;
    const float* images = (const float*)d_in[0];
    const float* w1     = (const float*)d_in[1];
    const float* b1     = (const float*)d_in[2];
    const float* w2     = (const float*)d_in[3];
    const float* b2     = (const float*)d_in[4];
    const float* w3     = (const float*)d_in[5];
    const float* b3     = (const float*)d_in[6];
    const float* wd     = (const float*)d_in[7];
    const float* bd     = (const float*)d_in[8];
    const float* c_x    = (const float*)d_in[9];
    const float* c_y    = (const float*)d_in[10];
    const float* comp_w = (const float*)d_in[11];
    const float* sigma  = (const float*)d_in[12];

    unsigned char* ws = (unsigned char*)d_ws;
    _Float16* w1T  = (_Float16*)(ws + OFF_W1T);
    _Float16* w2T  = (_Float16*)(ws + OFF_W2T);
    _Float16* w3T  = (_Float16*)(ws + OFF_W3T);
    _Float16* wdT  = (_Float16*)(ws + OFF_WDT);
    _Float16* cxT  = (_Float16*)(ws + OFF_CXT);
    float*    cxn2 = (float*)  (ws + OFF_CXN2);
    float*    yacc = (float*)  (ws + OFF_YACC);

    prep_kernel<<<128, 256, 0, stream>>>(w1, w2, w3, wd, c_x,
                                         w1T, w2T, w3T, wdT, cxT, cxn2, yacc);

    dim3 grid(GPI, BIMG);
    fused_kernel<<<grid, TPB, 0, stream>>>(images, b1, b2, b3, bd, comp_w, sigma,
                                           w1T, w2T, w3T, wdT, cxT, cxn2, yacc);

    finish_kernel<<<1, 192, 0, stream>>>(c_y, yacc, (float*)d_out);
}
